// MultiQueryDotProductAttention_73839077753206
// MI455X (gfx1250) — compile-verified, matrix operand order checked by reading, not run on hardware
//
#include <hip/hip_runtime.h>
#include <stdint.h>

// MQA attention block for MI455X (gfx1250, wave32, WMMA + TDM).
// B=2, LQ=LKV=2048, E=2048, H=16, DH=128.

typedef __bf16 bf16_t;
typedef bf16_t v16bf __attribute__((ext_vector_type(16)));
typedef float  v8f   __attribute__((ext_vector_type(8)));
typedef unsigned int v4u __attribute__((ext_vector_type(4)));
typedef int v8i __attribute__((ext_vector_type(8)));
typedef int v4i __attribute__((ext_vector_type(4)));

#define B_   2
#define LQ_  2048
#define LKV_ 2048
#define E_   2048
#define H_   16
#define DH_  128
#define HD_  (H_*DH_)   // 2048
#define KC_  128        // GEMM K-chunk staged per TDM transfer

static __device__ __forceinline__ unsigned short f32_to_bf16_rne(float f) {
    union { float f; uint32_t u; } v; v.f = f;
    uint32_t u = v.u;
    u += 0x7fffu + ((u >> 16) & 1u);   // round-to-nearest-even
    return (unsigned short)(u >> 16);
}

#define WMMA_BF16(a, b, c) \
    __builtin_amdgcn_wmma_f32_16x16x32_bf16(false, (a), false, (b), (short)0, (c), false, false)

// ---------------------------------------------------------------------------
// Tensor Data Mover: async 2D bf16 tile load Global -> LDS.
// D# per CDNA5 ISA:
//   group0: count=1 | lds_addr | global_addr(57b) | type=2
//   group1: data_size=1(2B), tensor_dim0=tile_k, tensor_dim1=tile_n,
//           tile_dim0=tile_k, tile_dim1=tile_n, dim0_stride=row_stride
//   groups2/3: zero (2D tensor)
// ---------------------------------------------------------------------------
static __device__ __forceinline__ void tdm_load_2d_bf16(
    const unsigned short* gsrc, unsigned int lds_off_bytes,
    unsigned int tile_k_elems, unsigned int tile_n_rows,
    unsigned long long row_stride_elems) {
    unsigned long long ga = (unsigned long long)(uintptr_t)gsrc;
    v4u g0;
    g0.x = 1u;                                           // count=1 (user mode)
    g0.y = lds_off_bytes;                                // lds_addr
    g0.z = (unsigned int)(ga & 0xFFFFFFFFu);             // global_addr[31:0]
    g0.w = (unsigned int)((ga >> 32) & 0x01FFFFFFu)      // global_addr[56:32]
         | (2u << 30);                                   // type=2 ("image")
    v8i g1;
    g1[0] = (int)(1u << 16);                             // data_size=1 (2 bytes)
    g1[1] = (int)((tile_k_elems & 0xFFFFu) << 16);       // tensor_dim0[15:0]
    g1[2] = (int)((tile_k_elems >> 16) |
                  ((tile_n_rows & 0xFFFFu) << 16));      // t_dim0 hi | t_dim1 lo
    g1[3] = (int)((tile_n_rows >> 16) |
                  ((tile_k_elems & 0xFFFFu) << 16));     // t_dim1 hi | tile_dim0
    g1[4] = (int)(tile_n_rows & 0xFFFFu);                // tile_dim1 (tile_dim2=0)
    g1[5] = (int)(row_stride_elems & 0xFFFFFFFFu);       // dim0_stride[31:0]
    g1[6] = (int)((row_stride_elems >> 32) & 0xFFFFu);   // dim0_stride[47:32]
    g1[7] = 0;
    v4i g2 = {0, 0, 0, 0};
    v4i g3 = {0, 0, 0, 0};
#if __clang_major__ >= 23
    v8i g4 = {0, 0, 0, 0, 0, 0, 0, 0};
    __builtin_amdgcn_tensor_load_to_lds(g0, g1, g2, g3, g4, 0);
#else
    __builtin_amdgcn_tensor_load_to_lds(g0, g1, g2, g3, 0);
#endif
}

// ---------------------------------------------------------------------------
// Elementwise f32 -> bf16 convert (optionally scaled).
// ---------------------------------------------------------------------------
__global__ void cvt_f32_to_bf16(const float* __restrict__ in,
                                unsigned short* __restrict__ out,
                                long long n, float scale) {
    long long i = (long long)blockIdx.x * blockDim.x + threadIdx.x;
    long long stride = (long long)gridDim.x * blockDim.x;
    for (; i < n; i += stride)
        out[i] = f32_to_bf16_rne(in[i] * scale);
}

// ---------------------------------------------------------------------------
// Transpose + convert: in[K][N] f32 -> out[N][K] bf16 (scaled).
// Makes WMMA B-operand fragment loads contiguous (lane = column N, 16
// contiguous K values per lane, the k order this code uses for both A and B).
// ---------------------------------------------------------------------------
__global__ void transpose_f32_to_bf16(const float* __restrict__ in,
                                      unsigned short* __restrict__ out,
                                      int K, int N, float scale) {
    long long n = (long long)K * N;
    long long i = (long long)blockIdx.x * blockDim.x + threadIdx.x;
    long long stride = (long long)gridDim.x * blockDim.x;
    for (; i < n; i += stride) {
        int kk = (int)(i / N);
        int nn = (int)(i % N);
        out[(long long)nn * K + kk] = f32_to_bf16_rne(in[i] * scale);
    }
}

// ---------------------------------------------------------------------------
// bf16 GEMM: C[M][N] = A[M][K] * Bt[N][K]^T, fp32 accumulate via WMMA.
// Block = 256 threads (8 waves); wave computes a 16x64 tile (4 C tiles).
// B panel (64 x KC bf16) is staged in LDS via the Tensor Data Mover,
// double-buffered: wave0 issues tensor_load_to_lds for chunk i+1 while all
// waves compute chunk i; one s_wait_tensorcnt + two barriers per chunk.
// A fragments stream from global (192MB L2 holds every operand here); the
// 4 A frags of a chunk are preloaded into distinct regs so global latency
// overlaps the WMMA+LDS pipeline instead of s_wait_loadcnt-serializing.
// ---------------------------------------------------------------------------
__global__ __launch_bounds__(256) void gemm_bf16_wmma(
    const unsigned short* __restrict__ A,    // [M][K] bf16
    const unsigned short* __restrict__ Bt,   // [N][K] bf16 (B transposed)
    float* __restrict__ Cf,                  // fp32 out (or null)
    unsigned short* __restrict__ Cb,         // bf16 out (or null)
    int M, int N, int K) {
    __shared__ __align__(128) unsigned short BtL[2][64 * KC_];  // 2 x 16KB

    const int lane = threadIdx.x & 31;
    const int wave = threadIdx.x >> 5;
    const int ln   = lane & 15;
    const int half = lane >> 4;

    const int m0 = blockIdx.y * 128 + wave * 16;
    const int n0 = blockIdx.x * 64;

    const unsigned short* arow = A + (size_t)(m0 + ln) * K + half * 16;
    const unsigned short* bpanel = Bt + (size_t)n0 * K;   // 64 rows of length K

    v8f c0 = {0,0,0,0,0,0,0,0};
    v8f c1 = c0, c2 = c0, c3 = c0;

    const int nchunks = K / KC_;
    // Prologue: TDM-stage chunk 0 into buffer 0.
    if (wave == 0) {
        tdm_load_2d_bf16(bpanel, (unsigned int)(uintptr_t)&BtL[0][0],
                         KC_, 64, (unsigned long long)K);
    }

    for (int c = 0; c < nchunks; ++c) {
        const int cur = c & 1;
        if (wave == 0) {
            __builtin_amdgcn_s_wait_tensorcnt(0);   // chunk c resident in LDS
        }
        __syncthreads();                            // publish chunk c
        if ((c + 1) < nchunks && wave == 0) {
            // Overlap: DMA chunk c+1 while everyone computes chunk c.
            tdm_load_2d_bf16(bpanel + (size_t)(c + 1) * KC_,
                             (unsigned int)(uintptr_t)&BtL[cur ^ 1][0],
                             KC_, 64, (unsigned long long)K);
        }

        const int kc = c * KC_;
        // Preload the chunk's 4 A fragments (distinct regs -> pipelined).
        v16bf a0 = *(const v16bf*)(arow + kc);
        v16bf a1 = *(const v16bf*)(arow + kc + 32);
        v16bf a2 = *(const v16bf*)(arow + kc + 64);
        v16bf a3 = *(const v16bf*)(arow + kc + 96);
        __builtin_prefetch(arow + kc + KC_, 0, 1);  // next chunk's A row

        const unsigned short* bl0 = &BtL[cur][(size_t)(ln) * KC_ + half * 16];
        const unsigned short* bl1 = &BtL[cur][(size_t)(16 + ln) * KC_ + half * 16];
        const unsigned short* bl2 = &BtL[cur][(size_t)(32 + ln) * KC_ + half * 16];
        const unsigned short* bl3 = &BtL[cur][(size_t)(48 + ln) * KC_ + half * 16];

        v16bf af[4] = {a0, a1, a2, a3};
#pragma unroll
        for (int j = 0; j < 4; ++j) {
            const int k2 = j * 32;
            v16bf b0 = *(const v16bf*)(bl0 + k2);
            v16bf b1 = *(const v16bf*)(bl1 + k2);
            v16bf b2 = *(const v16bf*)(bl2 + k2);
            v16bf b3 = *(const v16bf*)(bl3 + k2);
            c0 = WMMA_BF16(af[j], b0, c0);
            c1 = WMMA_BF16(af[j], b1, c1);
            c2 = WMMA_BF16(af[j], b2, c2);
            c3 = WMMA_BF16(af[j], b3, c3);
        }
        __syncthreads();   // all waves done with buffer `cur` before reuse
    }

    // C/D tile layout: VGPR r <-> row (r + 8*half), col = ln.
    const int colb = n0 + ln;
    v8f acc[4] = {c0, c1, c2, c3};
#pragma unroll
    for (int t = 0; t < 4; ++t) {
#pragma unroll
        for (int r = 0; r < 8; ++r) {
            int row = m0 + r + 8 * half;
            size_t off = (size_t)row * N + (colb + t * 16);
            if (Cf) Cf[off] = acc[t][r];
            else    Cb[off] = f32_to_bf16_rne(acc[t][r]);
        }
    }
}

// ---------------------------------------------------------------------------
// Flash-style MQA attention core.
// grid = (LQ/128, H, B); block = 256 threads = 8 waves; wave owns 16 q rows.
// kv chunks of 32: S = Q*K^T via WMMA (K rows are contiguous-in-d, i.e.
// natural B-fragments), online softmax with wave32 shfl_xor reductions,
// P relayout via per-wave LDS, O += P*V with V^T staged in LDS.
// ---------------------------------------------------------------------------
__global__ __launch_bounds__(256) void mqa_attn_wmma(
    const unsigned short* __restrict__ Qb,   // [B*LQ][HD] bf16 (pre-scaled)
    const unsigned short* __restrict__ Kb,   // [B*LKV][DH] bf16
    const unsigned short* __restrict__ Vb,   // [B*LKV][DH] bf16
    const float* __restrict__ bias,          // [B][1][LQ][LKV] f32
    unsigned short* __restrict__ Ctx) {      // [B*LQ][HD] bf16
    // V^T tile: [DH][48] (row stride 96B -> 32B-aligned frags, odd-ish banks)
    __shared__ __align__(128) unsigned short Vt[DH_ * 48];
    // Per-wave P scratch: 16x32 bf16
    __shared__ __align__(128) unsigned short Pl[8 * 16 * 32];

    const int tid  = threadIdx.x;
    const int wave = tid >> 5;
    const int lane = tid & 31;
    const int ln   = lane & 15;
    const int half = lane >> 4;

    const int b  = blockIdx.z;
    const int h  = blockIdx.y;
    const int q0 = blockIdx.x * 128 + wave * 16;

    const unsigned short* qbase = Qb + (size_t)b * LQ_ * HD_ + (size_t)h * DH_;
    const unsigned short* kbase = Kb + (size_t)b * LKV_ * DH_;
    const unsigned short* vbase = Vb + (size_t)b * LKV_ * DH_;
    const float*          bbase = bias + (size_t)b * LQ_ * LKV_;
    unsigned short*       pmine = &Pl[wave * 16 * 32];

    v8f O[8];
    v8f zero = {0,0,0,0,0,0,0,0};
#pragma unroll
    for (int j = 0; j < 8; ++j) O[j] = zero;

    float m[8], l[8];
#pragma unroll
    for (int r = 0; r < 8; ++r) { m[r] = -3.0e38f; l[r] = 0.0f; }

    const unsigned short* qrow = qbase + (size_t)(q0 + ln) * HD_ + half * 16;

    for (int kv0 = 0; kv0 < LKV_; kv0 += 32) {
        __syncthreads();   // protect previous chunk's Vt readers
        // Cooperative staging of V^T chunk: Vt[d][kv] = V[kv0+kv][d].
        for (int idx = tid; idx < 32 * DH_; idx += 256) {
            int kv = idx >> 7;       // 0..31
            int d  = idx & 127;      // 0..127
            Vt[d * 48 + kv] = vbase[(size_t)(kv0 + kv) * DH_ + d];
        }

        // ---- S = Q * K^T (two 16x16 tiles covering 32 kv columns) ----
        v8f s0 = zero, s1 = zero;
        const unsigned short* krow0 = kbase + (size_t)(kv0      + ln) * DH_ + half * 16;
        const unsigned short* krow1 = kbase + (size_t)(kv0 + 16 + ln) * DH_ + half * 16;
#pragma unroll
        for (int d0 = 0; d0 < DH_; d0 += 32) {
            v16bf a  = *(const v16bf*)(qrow  + d0);
            v16bf b0 = *(const v16bf*)(krow0 + d0);
            v16bf b1 = *(const v16bf*)(krow1 + d0);
            s0 = WMMA_BF16(a, b0, s0);
            s1 = WMMA_BF16(a, b1, s1);
        }
        __builtin_prefetch(krow0 + 32 * DH_, 0, 1);   // next chunk's K rows

        // ---- bias + online softmax (rows live across 16-lane halves) ----
        float alpha[8];
#pragma unroll
        for (int r = 0; r < 8; ++r) {
            const int qg = q0 + r + 8 * half;
            const float* brow = bbase + (size_t)qg * LKV_ + kv0;
            float x0 = s0[r] + brow[ln];
            float x1 = s1[r] + brow[16 + ln];
            float mx = fmaxf(x0, x1);
#pragma unroll
            for (int off = 1; off < 16; off <<= 1)
                mx = fmaxf(mx, __shfl_xor(mx, off, 32));
            float mn = fmaxf(m[r], mx);
            float al = __expf(m[r] - mn);
            float p0 = __expf(x0 - mn);
            float p1 = __expf(x1 - mn);
            float rs = p0 + p1;
#pragma unroll
            for (int off = 1; off < 16; off <<= 1)
                rs += __shfl_xor(rs, off, 32);
            m[r] = mn;
            l[r] = l[r] * al + rs;
            alpha[r] = al;
            const int lrow = r + 8 * half;
            pmine[lrow * 32 + ln]      = f32_to_bf16_rne(p0);
            pmine[lrow * 32 + 16 + ln] = f32_to_bf16_rne(p1);
        }
        // Wave-private LDS relayout: wait for our DS stores before A-frag read.
        asm volatile("s_wait_dscnt 0" ::: "memory");
        v16bf ap = *(const v16bf*)(pmine + ln * 32 + half * 16);

        __syncthreads();   // Vt chunk fully staged by all waves

        // ---- O = diag(alpha)*O + P * V ----
#pragma unroll
        for (int j = 0; j < 8; ++j) {
            v16bf bv = *(const v16bf*)(&Vt[(j * 16 + ln) * 48 + half * 16]);
            v8f o = O[j];
#pragma unroll
            for (int r = 0; r < 8; ++r) o[r] *= alpha[r];
            O[j] = WMMA_BF16(ap, bv, o);
        }
    }

    // ---- normalize and store ctx[b][q][h*DH + d] (bf16) ----
    float inv[8];
#pragma unroll
    for (int r = 0; r < 8; ++r) inv[r] = 1.0f / l[r];
#pragma unroll
    for (int j = 0; j < 8; ++j) {
#pragma unroll
        for (int r = 0; r < 8; ++r) {
            int qg = q0 + r + 8 * half;
            int d  = j * 16 + ln;
            Ctx[((size_t)b * LQ_ + qg) * HD_ + (size_t)h * DH_ + d] =
                f32_to_bf16_rne(O[j][r] * inv[r]);
        }
    }
}

// ---------------------------------------------------------------------------
// Host-side orchestration (all launches on `stream`, graph-capture safe).
// ---------------------------------------------------------------------------
extern "C" void kernel_launch(void* const* d_in, const int* in_sizes, int n_in,
                              void* d_out, int out_size, void* d_ws, size_t ws_size,
                              hipStream_t stream) {
    (void)in_sizes; (void)n_in; (void)out_size; (void)ws_size;

    const float* inq  = (const float*)d_in[0];   // [B, LQ, E]
    const float* inkv = (const float*)d_in[1];   // [B, LKV, E]
    const float* bias = (const float*)d_in[2];   // [B, 1, LQ, LKV]
    const float* Wq   = (const float*)d_in[3];   // [E, H, DH]
    const float* Wk   = (const float*)d_in[4];   // [E, DH]
    const float* Wv   = (const float*)d_in[5];   // [E, DH]
    const float* Wo   = (const float*)d_in[6];   // [H, DH, E]
    float* out = (float*)d_out;                  // [B, LQ, E] f32

    // Workspace layout (bf16 buffers, 256B aligned). ~87 MB total.
    char* ws = (char*)d_ws;
    unsigned short* Xq  = (unsigned short*)(ws + 0);            // [B*LQ][E]
    unsigned short* Xkv = (unsigned short*)(ws + 16777216);     // [B*LKV][E]
    unsigned short* WqT = (unsigned short*)(ws + 33554432);     // [HD][E]
    unsigned short* WkT = (unsigned short*)(ws + 41943040);     // [DH][E]
    unsigned short* WvT = (unsigned short*)(ws + 42467328);     // [DH][E]
    unsigned short* WoT = (unsigned short*)(ws + 42991616);     // [E][HD]
    unsigned short* Qb  = (unsigned short*)(ws + 51380224);     // [B*LQ][HD]
    unsigned short* Kb  = (unsigned short*)(ws + 68157440);     // [B*LKV][DH]
    unsigned short* Vb  = (unsigned short*)(ws + 69206016);     // [B*LKV][DH]
    unsigned short* Ctx = (unsigned short*)(ws + 70254592);     // [B*LQ][HD]

    const float qscale = 0.08838834764831845f;  // 1/sqrt(DH=128), folded into Wq

    // Stage 0: precision conversion (+ weight transposes for B-frag loads).
    cvt_f32_to_bf16<<<4096, 256, 0, stream>>>(inq,  Xq,  (long long)B_ * LQ_ * E_, 1.0f);
    cvt_f32_to_bf16<<<4096, 256, 0, stream>>>(inkv, Xkv, (long long)B_ * LKV_ * E_, 1.0f);
    transpose_f32_to_bf16<<<4096, 256, 0, stream>>>(Wq, WqT, E_, HD_, qscale);
    transpose_f32_to_bf16<<<512,  256, 0, stream>>>(Wk, WkT, E_, DH_, 1.0f);
    transpose_f32_to_bf16<<<512,  256, 0, stream>>>(Wv, WvT, E_, DH_, 1.0f);
    transpose_f32_to_bf16<<<4096, 256, 0, stream>>>(Wo, WoT, HD_, E_, 1.0f);

    // Stage 1: projections (WMMA GEMMs, bf16 out).
    dim3 gq(HD_ / 64, (B_ * LQ_) / 128);
    gemm_bf16_wmma<<<gq, 256, 0, stream>>>(Xq, WqT, nullptr, Qb, B_ * LQ_, HD_, E_);
    dim3 gkv(DH_ / 64, (B_ * LKV_) / 128);
    gemm_bf16_wmma<<<gkv, 256, 0, stream>>>(Xkv, WkT, nullptr, Kb, B_ * LKV_, DH_, E_);
    gemm_bf16_wmma<<<gkv, 256, 0, stream>>>(Xkv, WvT, nullptr, Vb, B_ * LKV_, DH_, E_);

    // Stage 2: flash attention core.
    dim3 ga(LQ_ / 128, H_, B_);
    mqa_attn_wmma<<<ga, 256, 0, stream>>>(Qb, Kb, Vb, bias, Ctx);

    // Stage 3: output projection (f32 out -> d_out).
    dim3 go(E_ / 64, (B_ * LQ_) / 128);
    gemm_bf16_wmma<<<go, 256, 0, stream>>>(Ctx, WoT, out, nullptr, B_ * LQ_, E_, HD_);
}
